// AdditiveAttention_8847632629895
// MI455X (gfx1250) — matrix operand order repaired and checked by reading, not compiled, not run on hardware
//
#include <hip/hip_runtime.h>
#include <hip/hip_bf16.h>
#include <math.h>

typedef __attribute__((ext_vector_type(16))) __bf16 v16bf;
typedef __attribute__((ext_vector_type(8)))  float  v8f;

static __device__ __forceinline__ unsigned short f2bf(float f) {
  union { float f; unsigned u; } x; x.f = f;
  unsigned r = x.u + 0x7FFFu + ((x.u >> 16) & 1u);   // round-to-nearest-even
  return (unsigned short)(r >> 16);
}

static __device__ __forceinline__ float dev_tanh(float x) {
#if __has_builtin(__builtin_amdgcn_tanhf)
  return __builtin_amdgcn_tanhf(x);                  // V_TANH_F32 (TRANS op)
#else
  return tanhf(x);
#endif
}

__global__ void cvt_f32_to_bf16(const float* __restrict__ src,
                                unsigned short* __restrict__ dst, int n) {
  int i = blockIdx.x * blockDim.x + threadIdx.x;
  int stride = gridDim.x * blockDim.x;
  for (; i < n; i += stride) dst[i] = f2bf(src[i]);
}

// Tiled transpose + f32->bf16: src [batch][R][Ccols] -> dst [batch][Ccols][R]
__global__ void transpose_cvt_bf16(const float* __restrict__ src,
                                   unsigned short* __restrict__ dst,
                                   int R, int Ccols) {
  __shared__ float tile[32][33];                     // +1 pad: no bank conflicts
  const long bofs = (long)blockIdx.z * R * Ccols;
  src += bofs; dst += bofs;
  const int r0 = blockIdx.y * 32, c0 = blockIdx.x * 32;
  const int tx = threadIdx.x, ty = threadIdx.y;      // 32 x 8
#pragma unroll
  for (int i = 0; i < 4; ++i)
    tile[ty + i * 8][tx] = src[(long)(r0 + ty + i * 8) * Ccols + c0 + tx];
  __syncthreads();
#pragma unroll
  for (int i = 0; i < 4; ++i)
    dst[(long)(c0 + ty + i * 8) * R + r0 + tx] = f2bf(tile[tx][ty + i * 8]);
}

// One wave computes a 16x64 fp32 tile of C = A(bf16,row-major[M,K]) x B.
// B stored K-contiguous per output column: B(k,n) = Bm[n*ldb + k]
// (weights are [N,K] natively; values are pre-transposed to [N,K]).
__global__ void gemm_bf16_wmma(const unsigned short* __restrict__ A,
                               const unsigned short* __restrict__ Bm,
                               float* __restrict__ C,
                               int M, int N, int Kd,
                               int lda, int ldb, int ldc,
                               long sA, long sB, long sC) {
  const int batch = blockIdx.y;
  A  += (long)batch * sA;
  Bm += (long)batch * sB;
  C  += (long)batch * sC;

  const int wavesPerBlock = blockDim.x >> 5;
  const int waveId = threadIdx.x >> 5;
  const int lane   = threadIdx.x & 31;
  const int tilesN = N >> 6;                         // 64-wide N tiles
  const int tile   = blockIdx.x * wavesPerBlock + waveId;
  const int tm = (tile / tilesN) << 4;
  const int tn = (tile % tilesN) << 6;
  if (tm >= M) return;                               // wave-uniform branch

  const int half  = lane >> 4;                       // 0: lanes 0-15, 1: 16-31
  const int l15   = lane & 15;
  const int koffA = half * 16;                       // A K-groups per half-wave
  const int koffB = half * 16;                       // B K-groups per half-wave

  v8f acc[4];
#pragma unroll
  for (int t = 0; t < 4; ++t) acc[t] = (v8f){0.f,0.f,0.f,0.f,0.f,0.f,0.f,0.f};

  union Frag { v16bf v; unsigned short s[16]; };

  const unsigned short* arow = A + (long)(tm + l15) * lda;
  const unsigned short* bcol[4];
#pragma unroll
  for (int t = 0; t < 4; ++t)
    bcol[t] = Bm + (long)(tn + 16 * t + l15) * ldb;

  for (int kk = 0; kk < Kd; kk += 32) {
    Frag fa;
    // A: VGPR j holds K pair koffA+2j (16 contiguous K per half-wave)
#pragma unroll
    for (int j = 0; j < 8; ++j) {
      int base = kk + koffA + 2 * j;
      fa.s[2 * j]     = arow[base];
      fa.s[2 * j + 1] = arow[base + 1];
    }
    // 4 independent WMMAs reuse the A fragment (16x64 per wave)
#pragma unroll
    for (int t = 0; t < 4; ++t) {
      Frag fb;
#pragma unroll
      for (int e = 0; e < 16; ++e)                   // 16 contiguous K -> b128
        fb.s[e] = bcol[t][kk + koffB + e];
      acc[t] = __builtin_amdgcn_wmma_f32_16x16x32_bf16(
          false, fa.v, false, fb.v, (short)0, acc[t], false, false);
    }
  }

  // C/D layout: VGPR i -> M = tm + i + 8*half, N = col + (lane&15)
#pragma unroll
  for (int t = 0; t < 4; ++t) {
#pragma unroll
    for (int i = 0; i < 8; ++i) {
      int row = tm + i + half * 8;
      C[(long)row * ldc + tn + 16 * t + l15] = acc[t][i];
    }
  }
}

// Fused: scores[b,q,k] = sum_h Wv[h]*tanh(qp[h]+kp[h]) -> mask -> softmax -> bf16
__global__ void attn_fused_kernel(const float* __restrict__ qproj,
                                  const float* __restrict__ kproj,
                                  const float* __restrict__ Wv,
                                  const int* __restrict__ valid_lens,
                                  unsigned short* __restrict__ attn_bf,
                                  int Qn, int Kn, int Hn) {
  __shared__ float qrow[128];
  __shared__ float wv[128];
  __shared__ float srow[1024];
  __shared__ float red[256];

  const int row = blockIdx.x;                        // b*Q + q
  const int b = row / Qn;
  if ((int)threadIdx.x < Hn) {
    qrow[threadIdx.x] = qproj[(long)row * Hn + threadIdx.x];
    wv[threadIdx.x]   = Wv[threadIdx.x];
  }
  __syncthreads();

  const int vlen   = valid_lens[b];
  const int wave   = threadIdx.x >> 5;
  const int lane   = threadIdx.x & 31;
  const int nwaves = blockDim.x >> 5;

  const float4 qq  = *(const float4*)(qrow + lane * 4);
  const float4 wv4 = *(const float4*)(wv + lane * 4);

  for (int k = wave; k < Kn; k += nwaves) {          // one wave per key row
    const float* krow = kproj + ((long)b * Kn + k) * Hn;
    const float4 kq = *(const float4*)(krow + lane * 4);   // coalesced b128
    __builtin_prefetch(krow + (long)nwaves * Hn + lane * 4, 0, 3);
    float s = wv4.x * dev_tanh(qq.x + kq.x)
            + wv4.y * dev_tanh(qq.y + kq.y)
            + wv4.z * dev_tanh(qq.z + kq.z)
            + wv4.w * dev_tanh(qq.w + kq.w);
#pragma unroll
    for (int off = 16; off > 0; off >>= 1)           // wave32 reduction
      s += __shfl_xor(s, off, 32);
    if (lane == 0) srow[k] = (k < vlen) ? s : -1e6f;
  }
  __syncthreads();

  // row softmax over srow[0..Kn)
  float m = -3.0e38f;
  for (int k = threadIdx.x; k < Kn; k += blockDim.x) m = fmaxf(m, srow[k]);
  red[threadIdx.x] = m; __syncthreads();
  for (int s = blockDim.x >> 1; s > 0; s >>= 1) {
    if ((int)threadIdx.x < s)
      red[threadIdx.x] = fmaxf(red[threadIdx.x], red[threadIdx.x + s]);
    __syncthreads();
  }
  m = red[0]; __syncthreads();

  float sum = 0.f;
  for (int k = threadIdx.x; k < Kn; k += blockDim.x) sum += __expf(srow[k] - m);
  red[threadIdx.x] = sum; __syncthreads();
  for (int s = blockDim.x >> 1; s > 0; s >>= 1) {
    if ((int)threadIdx.x < s) red[threadIdx.x] += red[threadIdx.x + s];
    __syncthreads();
  }
  const float inv = 1.f / red[0];

  for (int k = threadIdx.x; k < Kn; k += blockDim.x)
    attn_bf[(long)row * Kn + k] = f2bf(__expf(srow[k] - m) * inv);
}

extern "C" void kernel_launch(void* const* d_in, const int* in_sizes, int n_in,
                              void* d_out, int out_size, void* d_ws, size_t ws_size,
                              hipStream_t stream) {
  const int Bn = 16, Qn = 128, Kn = 1024, QS = 512, KS = 512, Hn = 128, DV = 512;

  const float* queries = (const float*)d_in[0];
  const float* keys    = (const float*)d_in[1];
  const float* values  = (const float*)d_in[2];
  const float* Wq      = (const float*)d_in[3];
  const float* Wk      = (const float*)d_in[4];
  const float* Wv      = (const float*)d_in[5];
  const int*   vlens   = (const int*)d_in[6];
  float* out = (float*)d_out;

  char* ws = (char*)d_ws;
  size_t off = 0;
  auto alloc = [&](size_t bytes) -> void* {
    void* p = ws + off;
    off = (off + bytes + 255) & ~(size_t)255;
    return p;
  };

  unsigned short* q_bf    = (unsigned short*)alloc((size_t)Bn * Qn * QS * 2);
  unsigned short* k_bf    = (unsigned short*)alloc((size_t)Bn * Kn * KS * 2);
  unsigned short* vT_bf   = (unsigned short*)alloc((size_t)Bn * Kn * DV * 2);
  unsigned short* wq_bf   = (unsigned short*)alloc((size_t)Hn * QS * 2);
  unsigned short* wk_bf   = (unsigned short*)alloc((size_t)Hn * KS * 2);
  float*          qproj   = (float*)alloc((size_t)Bn * Qn * Hn * 4);
  float*          kproj   = (float*)alloc((size_t)Bn * Kn * Hn * 4);
  unsigned short* attn_bf = (unsigned short*)alloc((size_t)Bn * Qn * Kn * 2);

  const int T = 256;
  auto nblk = [](int n, int t) { return (n + t - 1) / t; };

  // fp32 -> bf16 converts (weights + activations)
  cvt_f32_to_bf16<<<nblk(Bn*Qn*QS, T), T, 0, stream>>>(queries, q_bf, Bn*Qn*QS);
  cvt_f32_to_bf16<<<nblk(Bn*Kn*KS, T), T, 0, stream>>>(keys,    k_bf, Bn*Kn*KS);
  cvt_f32_to_bf16<<<nblk(Hn*QS, T),    T, 0, stream>>>(Wq,      wq_bf, Hn*QS);
  cvt_f32_to_bf16<<<nblk(Hn*KS, T),    T, 0, stream>>>(Wk,      wk_bf, Hn*KS);

  // values [B,K,DV] -> vT_bf [B,DV,K]  (K-contiguous per output column)
  {
    dim3 grid(DV / 32, Kn / 32, Bn), blk(32, 8);
    transpose_cvt_bf16<<<grid, blk, 0, stream>>>(values, vT_bf, Kn, DV);
  }

  // qproj = queries x Wq^T : M=2048, N=128 -> 128*2 wave-tiles / 8 waves
  {
    dim3 grid((Bn * Qn / 16) * (Hn / 64) / 8, 1);
    gemm_bf16_wmma<<<grid, T, 0, stream>>>(q_bf, wq_bf, qproj,
        Bn * Qn, Hn, QS, QS, QS, Hn, 0, 0, 0);
  }
  // kproj = keys x Wk^T : M=16384, N=128 -> 1024*2 wave-tiles
  {
    dim3 grid((Bn * Kn / 16) * (Hn / 64) / 8, 1);
    gemm_bf16_wmma<<<grid, T, 0, stream>>>(k_bf, wk_bf, kproj,
        Bn * Kn, Hn, KS, KS, KS, Hn, 0, 0, 0);
  }

  // fused tanh scores + mask + softmax -> bf16 attention
  attn_fused_kernel<<<Bn * Qn, T, 0, stream>>>(qproj, kproj, Wv, vlens,
                                               attn_bf, Qn, Kn, Hn);

  // out[b] = attn[b] x values[b] : per batch M=128, N=512 -> 8*8 wave-tiles
  {
    dim3 grid((Qn / 16) * (DV / 64) / 8, Bn);
    gemm_bf16_wmma<<<grid, T, 0, stream>>>(attn_bf, vT_bf, out,
        Qn, DV, Kn, Kn, Kn, DV,
        (long)Qn * Kn, (long)Kn * DV, (long)Qn * DV);
  }
}
